// Attention_180388626511
// MI455X (gfx1250) — matrix operand order repaired and checked by reading, not compiled, not run on hardware
//
#include <hip/hip_runtime.h>
#include <hip/hip_bf16.h>

// MI455X / gfx1250 fused causal attention (compile-only target).
// - All four matmuls on v_wmma_f32_16x16x32_bf16.
// - Global->LDS tile staging via Tensor Data Mover (tensor_load_to_lds +
//   s_wait_tensorcnt), double-buffered in the attention hot loop.

#define B_  2
#define S_  2048
#define D_  1024
#define H_  16
#define DH_ 64

typedef __bf16 bf16_t;
typedef __attribute__((ext_vector_type(16))) __bf16 v16bf;
typedef __attribute__((ext_vector_type(8)))  float  v8f;
typedef __attribute__((ext_vector_type(4)))  unsigned int v4u;
typedef __attribute__((ext_vector_type(8)))  int v8i;
typedef __attribute__((ext_vector_type(4)))  int v4i;

__device__ inline v8f zero8() {
    v8f z;
#pragma unroll
    for (int i = 0; i < 8; ++i) z[i] = 0.0f;
    return z;
}

__device__ inline v8f wmma_bf16(v16bf a, v16bf b, v8f c) {
    return __builtin_amdgcn_wmma_f32_16x16x32_bf16(false, a, false, b, (short)0, c,
                                                   false, false);
}

// Generic LDS pointer -> 32-bit LDS byte offset (low 32 address bits).
__device__ inline unsigned lds_off(const void* p) {
    return (unsigned)(unsigned long long)p;
}

// TDM 2D tile load: tile_w contiguous bf16 elements per row, tile_h rows with
// row stride `stride_elems`, packed linearly into LDS at lds_byte_addr.
// D# per ISA ch.8: group0 = {count, lds_addr, global_addr, type=2},
// group1 = {data_size=2B, tensor dims, tile dims, dim0 stride}. 2D => groups 2/3 zero.
// This toolchain exposes the 6-arg builtin (g0, g1, g2, g3, g4, cpol).
__device__ inline void tdm_load_2d_bf16(unsigned lds_byte_addr,
                                        const bf16_t* gptr,
                                        unsigned tile_w, unsigned tile_h,
                                        unsigned long long stride_elems) {
    unsigned long long ga = (unsigned long long)(const void*)gptr;
    v4u g0;
    g0[0] = 1u;                                                  // count=1, user D#
    g0[1] = lds_byte_addr;                                       // LDS addr (bytes)
    g0[2] = (unsigned)ga;                                        // global_addr[31:0]
    g0[3] = (unsigned)((ga >> 32) & 0x01FFFFFFu) | (2u << 30);   // [56:32] | type=2
    unsigned td0 = (unsigned)stride_elems;                       // tensor_dim0 >= tile_w
    unsigned td1 = tile_h;                                       // tensor_dim1
    v8i g1;
    g1[0] = (int)(1u << 16);                                     // wg_mask=0, data_size=1 (2B)
    g1[1] = (int)((td0 & 0xFFFFu) << 16);                        // tensor_dim0[15:0]
    g1[2] = (int)(((td0 >> 16) & 0xFFFFu) | ((td1 & 0xFFFFu) << 16));
    g1[3] = (int)(((td1 >> 16) & 0xFFFFu) | ((tile_w & 0xFFFFu) << 16)); // tile_dim0
    g1[4] = (int)(tile_h & 0xFFFFu);                             // tile_dim1, tile_dim2=0
    g1[5] = (int)(unsigned)(stride_elems & 0xFFFFFFFFull);       // dim0_stride[31:0]
    g1[6] = (int)(unsigned)((stride_elems >> 32) & 0xFFFFull);   // dim0_stride[47:32]
    g1[7] = 0;
    v4i z4;
    z4[0] = 0; z4[1] = 0; z4[2] = 0; z4[3] = 0;
    v8i z8;
#pragma unroll
    for (int i = 0; i < 8; ++i) z8[i] = 0;
    __builtin_amdgcn_tensor_load_to_lds(g0, g1, z4, z4, z8, 0);
}

// A-fragment (16x32 bf16, M x K): lane holds row M=lane&15.
// lanes 0-15: K = {0..7, 16..23}; lanes 16-31: K = {8..15, 24..31}.
__device__ inline v16bf load_frag_a(const bf16_t* base, int ldm) {
    const int lane = threadIdx.x & 31;
    const int m  = lane & 15;
    const int kb = (lane >> 4) * 8;
    const bf16_t* row = base + (size_t)m * ldm;
    v16bf f;
#pragma unroll
    for (int j = 0; j < 8; ++j) f[j]     = row[kb + j];
#pragma unroll
    for (int j = 0; j < 8; ++j) f[8 + j] = row[kb + 16 + j];
    return f;
}

// B-fragment (32x16 bf16, K x N) from N-major storage (element (k,n) at base[n*ldn+k]).
__device__ inline v16bf load_frag_b_t(const bf16_t* base, int ldn) {
    const int lane = threadIdx.x & 31;
    const int n  = lane & 15;
    const int kb = (lane >> 4) * 8;
    const bf16_t* row = base + (size_t)n * ldn;
    v16bf f;
#pragma unroll
    for (int j = 0; j < 8; ++j) f[j]     = row[kb + j];
#pragma unroll
    for (int j = 0; j < 8; ++j) f[8 + j] = row[kb + 16 + j];
    return f;
}

__global__ void f32_to_bf16_kernel(const float* __restrict__ in,
                                   bf16_t* __restrict__ out, int n) {
    int i = blockIdx.x * blockDim.x + threadIdx.x;
    if (i < n) out[i] = (bf16_t)in[i];
}

// ---------------------------------------------------------------------------
// QKV projection: OUT[b,h,s,e] = (X[b,s,:] @ W[h,:,:] + bias[h,:]) * scale
// X tile staged by TDM; W tile staged manually (needs N-major transpose).
// ---------------------------------------------------------------------------
template <int TRANSPOSE_V>
__global__ __launch_bounds__(256) void qkv_gemm_kernel(
    const bf16_t* __restrict__ X, const bf16_t* __restrict__ W,
    const float* __restrict__ bias, bf16_t* __restrict__ Out, float scale)
{
    __shared__ bf16_t lx[128 * 32];   // X tile [row][k] (TDM-filled)
    __shared__ bf16_t lw[64 * 32];    // W tile N-major [n][k]
    const int tid  = threadIdx.x;
    const int lane = tid & 31;
    const int wave = tid >> 5;
    const int sbase = blockIdx.x * 128;
    const int h = blockIdx.y;
    const int b = blockIdx.z;
    const bf16_t* Xg = X + ((size_t)(b * S_) + sbase) * D_;
    const bf16_t* Wg = W + (size_t)h * D_ * DH_;

    v8f acc[4];
#pragma unroll
    for (int t = 0; t < 4; ++t) acc[t] = zero8();

    for (int kc = 0; kc < D_; kc += 32) {
        if (tid < 32) { // lead wave: DMA the 128x32 X tile into LDS
            tdm_load_2d_bf16(lds_off(lx), Xg + kc, 32u, 128u,
                             (unsigned long long)D_);
            __builtin_amdgcn_s_wait_tensorcnt(0);
        }
        { // W tile: 32(k) x 64(n) -> transposed N-major (overlaps TDM issue)
            int idx0 = tid * 8;
            int k  = idx0 >> 6;
            int n0 = idx0 & 63;
            alignas(16) bf16_t tmp[8];
            *(uint4*)tmp = *(const uint4*)(Wg + (size_t)(kc + k) * DH_ + n0);
#pragma unroll
            for (int i = 0; i < 8; ++i) lw[(n0 + i) * 32 + k] = tmp[i];
        }
        __syncthreads();
        v16bf a = load_frag_a(&lx[wave * 16 * 32], 32);
#pragma unroll
        for (int t = 0; t < 4; ++t) {
            v16bf bf = load_frag_b_t(&lw[t * 16 * 32], 32);
            acc[t] = wmma_bf16(a, bf, acc[t]);
        }
        __syncthreads();
    }

    const int nloc = lane & 15, hi = lane >> 4;
#pragma unroll
    for (int t = 0; t < 4; ++t) {
#pragma unroll
        for (int r = 0; r < 8; ++r) {
            int s = sbase + wave * 16 + r + hi * 8;
            int e = t * 16 + nloc;
            float v = (acc[t][r] + bias[h * DH_ + e]) * scale;
            if (TRANSPOSE_V)
                Out[(((size_t)(b * H_ + h)) * DH_ + e) * S_ + s] = (bf16_t)v;
            else
                Out[(((size_t)(b * H_ + h)) * S_ + s) * DH_ + e] = (bf16_t)v;
        }
    }
}

// ---------------------------------------------------------------------------
// Flash-style causal attention. K/V tiles double-buffered through LDS by TDM:
// lead wave issues next tile pair, s_wait_tensorcnt(2) overlaps DMA with the
// current step's 8 WMMAs + online softmax. Trip count uniform per block.
// ---------------------------------------------------------------------------
__global__ __launch_bounds__(256) void attn_kernel(
    const bf16_t* __restrict__ Q, const bf16_t* __restrict__ K,
    const bf16_t* __restrict__ Vt, bf16_t* __restrict__ Z)
{
    __shared__ bf16_t lK[2][32 * 64];   // [key][e] row-major (TDM)
    __shared__ bf16_t lV[2][64 * 32];   // [e][key] row-major (TDM, from Vt)
    __shared__ bf16_t lp[8 * 16 * 32];  // per-wave P tile [16 q][32 k]
    const int tid  = threadIdx.x;
    const int lane = tid & 31;
    const int wave = tid >> 5;
    const int qbase = blockIdx.x * 128;
    const int h = blockIdx.y;
    const int b = blockIdx.z;
    const bf16_t* Qh = Q  + ((size_t)(b * H_ + h)) * S_ * DH_;
    const bf16_t* Kh = K  + ((size_t)(b * H_ + h)) * S_ * DH_;
    const bf16_t* Vh = Vt + ((size_t)(b * H_ + h)) * DH_ * S_;
    const int q0 = qbase + wave * 16;
    const bool lead = (tid < 32);

    v16bf aq[2];
    aq[0] = load_frag_a(Qh + (size_t)q0 * DH_,      DH_);
    aq[1] = load_frag_a(Qh + (size_t)q0 * DH_ + 32, DH_);

    v8f o[4];
#pragma unroll
    for (int t = 0; t < 4; ++t) o[t] = zero8();
    float mrow[8], lrow[8];
#pragma unroll
    for (int r = 0; r < 8; ++r) { mrow[r] = -1e30f; lrow[r] = 0.0f; }

    const int nloc = lane & 15, hi = lane >> 4;
    bf16_t* lpw = &lp[wave * 16 * 32];
    const int nsteps = (qbase + 128) / 32;   // causal upper bound, >= 4

    if (lead) { // preload step 0 tiles
        tdm_load_2d_bf16(lds_off(lK[0]), Kh, 64u, 32u, (unsigned long long)DH_);
        tdm_load_2d_bf16(lds_off(lV[0]), Vh, 32u, 64u, (unsigned long long)S_);
    }

    for (int i = 0; i < nsteps; ++i) {
        const int kb = i * 32;
        const int p = i & 1;
        if (lead) {
            if (i + 1 < nsteps) { // prefetch next pair into the other buffer
                tdm_load_2d_bf16(lds_off(lK[1 - p]), Kh + (size_t)(kb + 32) * DH_,
                                 64u, 32u, (unsigned long long)DH_);
                tdm_load_2d_bf16(lds_off(lV[1 - p]), Vh + (kb + 32),
                                 32u, 64u, (unsigned long long)S_);
                __builtin_amdgcn_s_wait_tensorcnt(2); // current pair landed
            } else {
                __builtin_amdgcn_s_wait_tensorcnt(0);
            }
        }
        __syncthreads(); // publish TDM-filled tiles to all waves
        const bf16_t* Kt = lK[p];
        const bf16_t* Vl = lV[p];

        // scores for 16q x 32k
        v8f c0 = zero8(), c1 = zero8();
#pragma unroll
        for (int ec = 0; ec < 2; ++ec) {
            v16bf bk0 = load_frag_b_t(Kt + ec * 32, 64);
            v16bf bk1 = load_frag_b_t(Kt + 16 * 64 + ec * 32, 64);
            c0 = wmma_bf16(aq[ec], bk0, c0);
            c1 = wmma_bf16(aq[ec], bk1, c1);
        }
#pragma unroll
        for (int r = 0; r < 8; ++r) {
            const int qi = q0 + r + hi * 8;
            float s0 = (kb + nloc      <= qi) ? c0[r] : -100000.0f;
            float s1 = (kb + 16 + nloc <= qi) ? c1[r] : -100000.0f;
            float red = fmaxf(s0, s1);
#pragma unroll
            for (int msk = 1; msk < 16; msk <<= 1)
                red = fmaxf(red, __shfl_xor(red, msk, 16));
            float mnew = fmaxf(mrow[r], red);
            float corr = __expf(mrow[r] - mnew);
            float p0 = __expf(s0 - mnew);
            float p1 = __expf(s1 - mnew);
            float rs = p0 + p1;
#pragma unroll
            for (int msk = 1; msk < 16; msk <<= 1)
                rs += __shfl_xor(rs, msk, 16);
            lrow[r] = lrow[r] * corr + rs;
            mrow[r] = mnew;
#pragma unroll
            for (int t = 0; t < 4; ++t) o[t][r] *= corr;
            const int mr = r + hi * 8;
            lpw[mr * 32 + nloc]      = (bf16_t)p0;
            lpw[mr * 32 + 16 + nloc] = (bf16_t)p1;
        }
        // wave-local LDS transpose of P; DS ops in-order per wave
        v16bf pa = load_frag_a(lpw, 32);
#pragma unroll
        for (int t = 0; t < 4; ++t) {
            v16bf bv = load_frag_b_t(Vl + (size_t)(t * 16) * 32, 32);
            o[t] = wmma_bf16(pa, bv, o[t]);
        }
        __syncthreads(); // all waves done with buf[p] before it is re-filled
    }

#pragma unroll
    for (int r = 0; r < 8; ++r) {
        float inv = 1.0f / lrow[r];
        int s = q0 + r + hi * 8;
#pragma unroll
        for (int t = 0; t < 4; ++t) {
            int col = h * DH_ + t * 16 + nloc;
            Z[((size_t)(b * S_ + s)) * D_ + col] = (bf16_t)(o[t][r] * inv);
        }
    }
}

// ---------------------------------------------------------------------------
// Out projection: Out[m, d] = Z[m, :] @ Wo_flat[:, d] + bO[d]  (f32 output)
// A tile staged by TDM; Wo tile staged manually (transpose to N-major).
// ---------------------------------------------------------------------------
__global__ __launch_bounds__(256) void out_proj_kernel(
    const bf16_t* __restrict__ A, const bf16_t* __restrict__ Wo,
    const float* __restrict__ bO, float* __restrict__ Out)
{
    __shared__ bf16_t la[128 * 32];
    __shared__ bf16_t lb[128 * 32];   // N-major [n][k]
    const int tid = threadIdx.x, lane = tid & 31, wave = tid >> 5;
    const int mbase = blockIdx.x * 128;
    const int nbase = blockIdx.y * 128;
    const bf16_t* Ag = A + (size_t)mbase * D_;

    v8f acc[8];
#pragma unroll
    for (int t = 0; t < 8; ++t) acc[t] = zero8();

    for (int kc = 0; kc < D_; kc += 32) {
        if (tid < 32) {
            tdm_load_2d_bf16(lds_off(la), Ag + kc, 32u, 128u,
                             (unsigned long long)D_);
            __builtin_amdgcn_s_wait_tensorcnt(0);
        }
        {
            int idx0 = tid * 16;
            int k  = idx0 >> 7;
            int n0 = idx0 & 127;
            alignas(16) bf16_t tmp[16];
            *(uint4*)(&tmp[0]) = *(const uint4*)(Wo + ((size_t)(kc + k)) * D_ + nbase + n0);
            *(uint4*)(&tmp[8]) = *(const uint4*)(Wo + ((size_t)(kc + k)) * D_ + nbase + n0 + 8);
#pragma unroll
            for (int i = 0; i < 16; ++i) lb[(n0 + i) * 32 + k] = tmp[i];
        }
        __syncthreads();
        v16bf a = load_frag_a(&la[wave * 16 * 32], 32);
#pragma unroll
        for (int t = 0; t < 8; ++t) {
            v16bf bf = load_frag_b_t(&lb[t * 16 * 32], 32);
            acc[t] = wmma_bf16(a, bf, acc[t]);
        }
        __syncthreads();
    }

    const int nloc = lane & 15, hi = lane >> 4;
#pragma unroll
    for (int t = 0; t < 8; ++t) {
#pragma unroll
        for (int r = 0; r < 8; ++r) {
            int m = mbase + wave * 16 + r + hi * 8;
            int n = nbase + t * 16 + nloc;
            Out[(size_t)m * D_ + n] = acc[t][r] + bO[n];
        }
    }
}

// ---------------------------------------------------------------------------
extern "C" void kernel_launch(void* const* d_in, const int* in_sizes, int n_in,
                              void* d_out, int out_size, void* d_ws, size_t ws_size,
                              hipStream_t stream) {
    (void)in_sizes; (void)n_in; (void)out_size; (void)ws_size;
    const float* residual = (const float*)d_in[0];
    const float* x   = (const float*)d_in[1];
    const float* W_Q = (const float*)d_in[2];
    const float* W_K = (const float*)d_in[3];
    const float* W_V = (const float*)d_in[4];
    const float* W_O = (const float*)d_in[5];
    const float* b_Q = (const float*)d_in[6];
    const float* b_K = (const float*)d_in[7];
    const float* b_V = (const float*)d_in[8];
    const float* b_O = (const float*)d_in[9];
    float* out = (float*)d_out;

    const int nx = B_ * S_ * D_;          // 4,194,304
    const int nw = H_ * D_ * DH_;         // 1,048,576
    const int nqkv = B_ * H_ * S_ * DH_;  // 4,194,304

    size_t off = 0;
    auto carve = [&](size_t bytes) {
        void* p = (char*)d_ws + off;
        off += (bytes + 255) & ~(size_t)255;
        return p;
    };
    bf16_t* xb  = (bf16_t*)carve((size_t)nx * 2);
    bf16_t* wqb = (bf16_t*)carve((size_t)nw * 2);
    bf16_t* wkb = (bf16_t*)carve((size_t)nw * 2);
    bf16_t* wvb = (bf16_t*)carve((size_t)nw * 2);
    bf16_t* wob = (bf16_t*)carve((size_t)nw * 2);
    bf16_t* Qb  = (bf16_t*)carve((size_t)nqkv * 2);
    bf16_t* Kb  = (bf16_t*)carve((size_t)nqkv * 2);
    bf16_t* Vtb = (bf16_t*)carve((size_t)nqkv * 2);
    bf16_t* Zb  = (bf16_t*)carve((size_t)nx * 2);

    (void)hipMemcpyAsync(out, residual, (size_t)nx * sizeof(float),
                         hipMemcpyDeviceToDevice, stream);

    f32_to_bf16_kernel<<<(nx + 255) / 256, 256, 0, stream>>>(x, xb, nx);
    f32_to_bf16_kernel<<<(nw + 255) / 256, 256, 0, stream>>>(W_Q, wqb, nw);
    f32_to_bf16_kernel<<<(nw + 255) / 256, 256, 0, stream>>>(W_K, wkb, nw);
    f32_to_bf16_kernel<<<(nw + 255) / 256, 256, 0, stream>>>(W_V, wvb, nw);
    f32_to_bf16_kernel<<<(nw + 255) / 256, 256, 0, stream>>>(W_O, wob, nw);

    dim3 g1(S_ / 128, H_, B_);
    qkv_gemm_kernel<0><<<g1, 256, 0, stream>>>(xb, wqb, b_Q, Qb, 0.125f);
    qkv_gemm_kernel<0><<<g1, 256, 0, stream>>>(xb, wkb, b_K, Kb, 1.0f);
    qkv_gemm_kernel<1><<<g1, 256, 0, stream>>>(xb, wvb, b_V, Vtb, 1.0f);

    attn_kernel<<<g1, 256, 0, stream>>>(Qb, Kb, Vtb, Zb);

    dim3 g2((B_ * S_) / 128, D_ / 128);
    out_proj_kernel<<<g2, 256, 0, stream>>>(Zb, wob, b_O, out + nx);
}
